// Freq_Self_Att_weight_87488483819798
// MI455X (gfx1250) — matrix operand order repaired and checked by reading, not compiled, not run on hardware
//
#include <hip/hip_runtime.h>
#include <math.h>

// Freq self-attention channel re-weighting, MI455X (gfx1250, wave32, WMMA).
// Key algebraic reduction: (att @ x).mean_t == att @ mean_t(x), so the
// [B,512,4000] GEMM collapses to a [512]-dot per row. Remaining HBM traffic:
// read x twice + write out once ~= 393 MB -> ~17us floor @ 23.3 TB/s.
// Small GEMMs (pooled@Wk^T, pooled@Wq^T, key@query^T) run on
// v_wmma_f32_16x16x32_f16 (f16 in, f32 accumulate).

typedef __attribute__((ext_vector_type(16))) _Float16 v16h;
typedef __attribute__((ext_vector_type(8)))  float    v8f;

#define B_   16
#define F_   512
#define T_   4000
#define P_   100      // pooled length
#define KP_  128      // P_ padded to multiple of 32 for WMMA K
#define N_   256      // key/query feature dim
#define ROWS (B_*F_)  // 8192

// ---- WMMA fragment loaders (layouts per CDNA5 ISA 7.12.2, wave32) ----------
// A: 16x32 f16, row m = lane%16; halves 0..7 -> K = k0 + (lane>=16?8:0) + h,
//    halves 8..15 -> K = k0+16 + (lane>=16?8:0) + (h-8). Two 16B loads.
__device__ __forceinline__ v16h ldA(const _Float16* A, int ld, int m0, int k0) {
  int lane = threadIdx.x & 31;
  const _Float16* p = A + (size_t)(m0 + (lane & 15)) * ld + k0 + ((lane >> 4) << 3);
  v16h a;
  ((float4*)&a)[0] = *(const float4*)p;
  ((float4*)&a)[1] = *(const float4*)(p + 16);
  return a;
}
// B: 32x16 f16 stored as row-major [N][K]; col n = lane%16,
//    halves 0..7 -> K = k0 + (lane>=16?8:0) + h, halves 8..15 -> K = k0+16 + (lane>=16?8:0) + (h-8).
__device__ __forceinline__ v16h ldB(const _Float16* W, int ld, int n0, int k0) {
  int lane = threadIdx.x & 31;
  const _Float16* p = W + (size_t)(n0 + (lane & 15)) * ld + k0 + ((lane >> 4) << 3);
  v16h b;
  ((float4*)&b)[0] = *(const float4*)p;
  ((float4*)&b)[1] = *(const float4*)(p + 16);
  return b;
}

// ---- 0: convert Wk/Wq [256,100] f32 -> f16 [256][128] zero-padded ----------
__global__ void prep_weights(const float* __restrict__ Wk, const float* __restrict__ Wq,
                             _Float16* __restrict__ Wk16, _Float16* __restrict__ Wq16) {
  int i = blockIdx.x * blockDim.x + threadIdx.x;   // 0 .. 256*128-1
  int n = i >> 7, k = i & (KP_ - 1);
  Wk16[i] = (_Float16)((k < P_) ? Wk[n * P_ + k] : 0.f);
  Wq16[i] = (_Float16)((k < P_) ? Wq[n * P_ + k] : 0.f);
}

// ---- 1: adaptive avg pool (windows of 40) + per-row mean -------------------
__global__ void pool_kernel(const float* __restrict__ x,
                            _Float16* __restrict__ pooledH, float* __restrict__ xmean) {
  __shared__ float red[128];
  int row = blockIdx.x;           // 0..8191
  int p = threadIdx.x;            // 0..127
  float s = 0.f;
  if (p < P_) {
    const float* xr = x + (size_t)row * T_ + p * 40;
#pragma unroll
    for (int i = 0; i < 40; ++i) s += xr[i];
  }
  pooledH[(size_t)row * KP_ + p] = (p < P_) ? (_Float16)(s * (1.f / 40.f)) : (_Float16)0.f;
  red[p] = s;
  __syncthreads();
  for (int off = 64; off; off >>= 1) {
    if (p < off) red[p] += red[p + off];
    __syncthreads();
  }
  if (p == 0) xmean[row] = red[0] * (1.f / (float)T_);
}

// ---- 2: key/query = pooled @ W^T + b, one wave per 16x16 tile, WMMA --------
__global__ void kq_gemm(const _Float16* __restrict__ pooledH,
                        const _Float16* __restrict__ Wk16, const _Float16* __restrict__ Wq16,
                        const float* __restrict__ bk, const float* __restrict__ bq,
                        _Float16* __restrict__ key16, _Float16* __restrict__ qry16) {
  int wave = (blockIdx.x * blockDim.x + threadIdx.x) >> 5;  // 0..8191
  int lane = threadIdx.x & 31;
  int mt = wave >> 4;               // 512 row tiles over ROWS
  int n0 = (wave & 15) << 4;        // 16 col tiles over N_
  v8f ak = {}, aq = {};
#pragma unroll
  for (int kk = 0; kk < 4; ++kk) {  // K = 128
    v16h a   = ldA(pooledH, KP_, mt * 16, kk * 32);
    v16h bkf = ldB(Wk16, KP_, n0, kk * 32);
    v16h bqf = ldB(Wq16, KP_, n0, kk * 32);
    ak = __builtin_amdgcn_wmma_f32_16x16x32_f16(false, a, false, bkf, (short)0, ak, false, false);
    aq = __builtin_amdgcn_wmma_f32_16x16x32_f16(false, a, false, bqf, (short)0, aq, false, false);
  }
  int n = n0 + (lane & 15);
  float bkv = bk[n], bqv = bq[n];
  int mbase = mt * 16 + ((lane >> 4) << 3);
#pragma unroll
  for (int r = 0; r < 8; ++r) {     // C layout: col n fixed per lane, rows stride 1
    key16[(size_t)(mbase + r) * N_ + n] = (_Float16)(ak[r] + bkv);
    qry16[(size_t)(mbase + r) * N_ + n] = (_Float16)(aq[r] + bqv);
  }
}

// ---- 3: S = key@query^T /16, softmax rows fused with xmean dot -> fw -------
__global__ void att_fw(const _Float16* __restrict__ key16, const _Float16* __restrict__ qry16,
                       const float* __restrict__ xmean, float* __restrict__ fw) {
  __shared__ float sm[16 * F_];   // 32 KB scaled scores for this 16-row tile
  __shared__ float xm[F_];
  int b  = blockIdx.x >> 5;       // 16 batches
  int mt = blockIdx.x & 31;       // 32 row tiles of 16
  int tid = threadIdx.x, lane = tid & 31, w = tid >> 5;   // 8 waves
  const _Float16* Kb = key16 + (size_t)b * F_ * N_;
  const _Float16* Qb = qry16 + (size_t)b * F_ * N_;
  xm[tid]       = xmean[b * F_ + tid];
  xm[tid + 256] = xmean[b * F_ + tid + 256];

  v16h afr[8];                    // hoist A fragments (shared across col tiles)
#pragma unroll
  for (int kk = 0; kk < 8; ++kk) afr[kk] = ldA(Kb, N_, mt * 16, kk * 32);

  for (int c4 = 0; c4 < 4; ++c4) {       // each wave: 4 column tiles (64 cols)
    int ct = w * 4 + c4;
    v8f acc = {};
#pragma unroll
    for (int kk = 0; kk < 8; ++kk) {     // K = 256
      v16h bf = ldB(Qb, N_, ct * 16, kk * 32);
      acc = __builtin_amdgcn_wmma_f32_16x16x32_f16(false, afr[kk], false, bf, (short)0, acc, false, false);
    }
    int col = ct * 16 + (lane & 15);
    int rbase = (lane >> 4) << 3;
#pragma unroll
    for (int r = 0; r < 8; ++r) sm[(rbase + r) * F_ + col] = acc[r] * 0.0625f; // /sqrt(256)
  }
  __syncthreads();

  // softmax over 512 cols + dot with xmean; wave w handles rows 2w, 2w+1
#pragma unroll
  for (int rr = 0; rr < 2; ++rr) {
    int row = w * 2 + rr;
    const float* srow = sm + row * F_;
    float m = -INFINITY;
    for (int i = lane; i < F_; i += 32) m = fmaxf(m, srow[i]);
    for (int off = 16; off; off >>= 1) m = fmaxf(m, __shfl_xor(m, off, 32));
    float s1 = 0.f, s2 = 0.f;
    for (int i = lane; i < F_; i += 32) {
      float e = __expf(srow[i] - m);
      s1 += e; s2 += e * xm[i];
    }
    for (int off = 16; off; off >>= 1) {
      s1 += __shfl_xor(s1, off, 32);
      s2 += __shfl_xor(s2, off, 32);
    }
    if (lane == 0) fw[b * F_ + mt * 16 + row] = s2 / s1;
  }
}

// ---- 4: per-batch min/max normalize of fw ----------------------------------
__global__ void minmax_norm(const float* __restrict__ fw, float* __restrict__ fwn) {
  __shared__ float smin[256], smax[256];
  int b = blockIdx.x, tid = threadIdx.x;
  float v0 = fw[b * F_ + tid], v1 = fw[b * F_ + tid + 256];
  smin[tid] = fminf(v0, v1); smax[tid] = fmaxf(v0, v1);
  __syncthreads();
  for (int off = 128; off; off >>= 1) {
    if (tid < off) {
      smin[tid] = fminf(smin[tid], smin[tid + off]);
      smax[tid] = fmaxf(smax[tid], smax[tid + off]);
    }
    __syncthreads();
  }
  float mi = smin[0], ma = smax[0], inv = 1.f / (ma - mi);
  fwn[b * F_ + tid]       = (v0 - mi) * inv + 1e-6f;
  fwn[b * F_ + tid + 256] = (v1 - mi) * inv + 1e-6f;
}

// ---- 5: out = x * fw_n (HBM-bound pass, float4) ----------------------------
__global__ void scale_out(const float* __restrict__ x, const float* __restrict__ fwn,
                          float* __restrict__ out) {
  int row = blockIdx.x;
  float s = fwn[row];
  const float4* xr = (const float4*)(x + (size_t)row * T_);
  float4* orow = (float4*)(out + (size_t)row * T_);
  for (int i = threadIdx.x; i < T_ / 4; i += blockDim.x) {
    float4 v = xr[i];
    v.x *= s; v.y *= s; v.z *= s; v.w *= s;
    orow[i] = v;
  }
}

extern "C" void kernel_launch(void* const* d_in, const int* in_sizes, int n_in,
                              void* d_out, int out_size, void* d_ws, size_t ws_size,
                              hipStream_t stream) {
  const float* x  = (const float*)d_in[0];
  const float* Wk = (const float*)d_in[1];
  const float* bk = (const float*)d_in[2];
  const float* Wq = (const float*)d_in[3];
  const float* bq = (const float*)d_in[4];
  float* out = (float*)d_out;

  // workspace carve-out (~10.3 MB total)
  char* ws = (char*)d_ws;
  size_t off = 0;
  auto carve = [&](size_t bytes) { void* p = ws + off; off += (bytes + 255) & ~(size_t)255; return p; };
  _Float16* pooledH = (_Float16*)carve((size_t)ROWS * KP_ * sizeof(_Float16)); // 2 MB
  _Float16* Wk16    = (_Float16*)carve((size_t)N_ * KP_ * sizeof(_Float16));   // 64 KB
  _Float16* Wq16    = (_Float16*)carve((size_t)N_ * KP_ * sizeof(_Float16));   // 64 KB
  _Float16* key16   = (_Float16*)carve((size_t)ROWS * N_ * sizeof(_Float16));  // 4 MB
  _Float16* qry16   = (_Float16*)carve((size_t)ROWS * N_ * sizeof(_Float16));  // 4 MB
  float*    xmean   = (float*)carve((size_t)ROWS * sizeof(float));
  float*    fw      = (float*)carve((size_t)ROWS * sizeof(float));
  float*    fwn     = (float*)carve((size_t)ROWS * sizeof(float));

  prep_weights<<<128, 256, 0, stream>>>(Wk, Wq, Wk16, Wq16);
  pool_kernel <<<ROWS, 128, 0, stream>>>(x, pooledH, xmean);
  kq_gemm     <<<1024, 256, 0, stream>>>(pooledH, Wk16, Wq16, bk, bq, key16, qry16);
  att_fw      <<<B_ * 32, 256, 0, stream>>>(key16, qry16, xmean, fw);
  minmax_norm <<<B_, 256, 0, stream>>>(fw, fwn);
  scale_out   <<<ROWS, 256, 0, stream>>>(x, fwn, out);
  (void)in_sizes; (void)n_in; (void)out_size; (void)ws_size;
}
